// PolygonSegmenterGCN_11227044512214
// MI455X (gfx1250) — matrix operand order repaired and checked by reading, not compiled, not run on hardware
//
#include <hip/hip_runtime.h>
#include <hip/hip_bf16.h>

// CDNA5 / gfx1250, wave32. WMMA fragment types.
typedef __attribute__((ext_vector_type(16))) _Float16 v16h;
typedef __attribute__((ext_vector_type(8)))  float    v8f;

// ---------------------------------------------------------------------------
// Weight pre-pack: W[Cin x Cout] fp32 row-major  ->  fp16 in WMMA B-fragment
// order. Contiguous k order used by this code for the B (32x16) tile, same as the A fragment:
//   lane L holds column n = L%16, halves j=0..15 -> K = (L/16)*16 + j.
// Packed index: (((chunk*TILES + tile)*32 + lane)*16 + j), chunk = K/32.
// Each lane of the GEMM then loads 32 contiguous bytes (b128 pair).
// ---------------------------------------------------------------------------
__global__ void pack_w_f16(const float* __restrict__ W, _Float16* __restrict__ Bpk,
                           int Cout, int tiles, int total) {
  int idx = blockIdx.x * blockDim.x + threadIdx.x;
  if (idx >= total) return;
  int j     = idx & 15;
  int lane  = (idx >> 4) & 31;
  int rest  = idx >> 9;
  int tile  = rest % tiles;
  int chunk = rest / tiles;
  int K = chunk * 32 + (lane >> 4) * 16 + j;
  int n = tile * 16 + (lane & 15);
  Bpk[idx] = (_Float16)W[(size_t)K * Cout + n];
}

// ---------------------------------------------------------------------------
// GEMM: H[N x Cout] = X[N x Cin] * W(packed)   one wave per 16-row strip,
// computing ALL Cout columns: A fragment loaded once per 32-wide K chunk and
// reused across TILES column tiles (8x less A traffic than tile-per-wave).
// CHUNKS = Cin/32, TILES = Cout/16 as compile-time constants -> fully
// unrolled CHUNKS*TILES v_wmma_f32_16x16x32_f16 chain.
// ---------------------------------------------------------------------------
template <int CHUNKS, int TILES>
__global__ void gcn_gemm_wmma(const float* __restrict__ X,
                              const _Float16* __restrict__ Bpk,
                              float* __restrict__ H, int nStrips) {
  const int lane    = threadIdx.x & 31;
  const int strip   = blockIdx.x * (blockDim.x >> 5) + (threadIdx.x >> 5);
  if (strip >= nStrips) return;                // wave-uniform; EXEC stays all-1s
  const int Cin     = CHUNKS * 32;
  const int Cout    = TILES * 16;
  const int rowBase = strip * 16;
  const int m       = lane & 15;               // A row within tile / output col%16
  const int halfSel = lane >> 4;               // lane half select

  v8f acc[TILES];
#pragma unroll
  for (int t = 0; t < TILES; ++t) acc[t] = v8f{};

#pragma unroll
  for (int chunk = 0; chunk < CHUNKS; ++chunk) {
    const int kk = chunk * 32;
    // A fragment (contiguous k order, same as packed B): lane L: M = L%16;
    //   halves 0..15 -> K = kk + (L/16)*16 + j
    v16h a;
    const float* arow = X + (size_t)(rowBase + m) * Cin + kk + halfSel * 16;
#pragma unroll
    for (int j = 0; j < 8; ++j) {
      a[j]     = (_Float16)arow[j];
      a[j + 8] = (_Float16)arow[8 + j];
    }
    // B fragments: contiguous 32B per lane, pre-packed & L2-resident.
#pragma unroll
    for (int t = 0; t < TILES; ++t) {
      const v16h b = *(const v16h*)(Bpk + (((size_t)(chunk * TILES + t) * 32 + lane) << 4));
      acc[t] = __builtin_amdgcn_wmma_f32_16x16x32_f16(
          /*neg_a=*/false, a, /*neg_b=*/false, b,
          /*c_mod=*/(short)0, acc[t], /*reuse_a=*/false, /*reuse_b=*/false);
    }
  }

  // C/D layout: VGPR r -> M = r + 8*(lane/16), N = lane%16
#pragma unroll
  for (int t = 0; t < TILES; ++t) {
    float* hrow = H + (size_t)(rowBase + halfSel * 8) * Cout + t * 16 + m;
#pragma unroll
    for (int r = 0; r < 8; ++r)
      hrow[(size_t)r * Cout] = acc[t][r];
  }
}

// ---------------------------------------------------------------------------
// Normalization setup
// ---------------------------------------------------------------------------
__global__ void zero_f32(float* __restrict__ p, int n) {
  int i = blockIdx.x * blockDim.x + threadIdx.x;
  if (i < n) p[i] = 0.0f;
}

__global__ void deg_accum(const int* __restrict__ col,
                          const float* __restrict__ ew,
                          float* __restrict__ deg, int E) {
  int e = blockIdx.x * blockDim.x + threadIdx.x;
  if (e < E) atomicAdd(&deg[col[e]], ew[e]);
}

// in-place: deg -> dinv ; self-loop adds +1 so deg >= 1 always
__global__ void deg_to_dinv(float* __restrict__ d, int n) {
  int i = blockIdx.x * blockDim.x + threadIdx.x;
  if (i < n) d[i] = rsqrtf(d[i] + 1.0f);
}

__global__ void edge_norm(const int* __restrict__ row, const int* __restrict__ col,
                          const float* __restrict__ ew, const float* __restrict__ dinv,
                          float* __restrict__ norm, int E) {
  int e = blockIdx.x * blockDim.x + threadIdx.x;
  if (e < E) norm[e] = dinv[row[e]] * ew[e] * dinv[col[e]];
}

// ---------------------------------------------------------------------------
// Aggregation: OUT starts as bias, then atomic scatter-add of norm*H[row] into
// OUT[col] (one wave per edge, lanes stride channels -> coalesced gather; the
// 51MB feature matrices are L2-resident on MI455X's 192MB L2, so atomics are
// serviced at L2, not HBM), then fused self-loop + ReLU epilogue.
// ---------------------------------------------------------------------------
__global__ void init_bias(float* __restrict__ OUT, const float* __restrict__ b,
                          size_t total, int cmask) {
  size_t i = (size_t)blockIdx.x * blockDim.x + threadIdx.x;
  if (i < total) OUT[i] = b[i & (size_t)cmask];
}

__global__ void scatter_edges(const int* __restrict__ row, const int* __restrict__ col,
                              const float* __restrict__ norm,
                              const float* __restrict__ H,
                              float* __restrict__ OUT, int E, int C) {
  int gw   = (blockIdx.x * blockDim.x + threadIdx.x) >> 5;  // wave = edge
  int lane = threadIdx.x & 31;
  if (gw >= E) return;
  int   r = row[gw];
  int   d = col[gw];
  float w = norm[gw];
  const float* hs = H   + (size_t)r * C;
  float*       od = OUT + (size_t)d * C;
  for (int ch = lane; ch < C; ch += 32)
    atomicAdd(&od[ch], w * hs[ch]);
}

__global__ void selfloop_epilogue(float* __restrict__ OUT, const float* __restrict__ H,
                                  const float* __restrict__ dinv,
                                  size_t total, int cshift, int doRelu) {
  size_t i = (size_t)blockIdx.x * blockDim.x + threadIdx.x;
  if (i >= total) return;
  int   node = (int)(i >> cshift);
  float di   = dinv[node];
  float v    = OUT[i] + di * di * H[i];
  OUT[i] = doRelu ? fmaxf(v, 0.0f) : v;
}

// ---------------------------------------------------------------------------
// Link decoder: one wave per pair, 64 channels = 2 per lane, shfl_xor reduce.
// ---------------------------------------------------------------------------
__global__ void decode_pairs(const float* __restrict__ enc,
                             const int* __restrict__ eli,
                             float* __restrict__ out, int P) {
  int gw   = (blockIdx.x * blockDim.x + threadIdx.x) >> 5;
  int lane = threadIdx.x & 31;
  if (gw >= P) return;
  int a = eli[gw];
  int b = eli[P + gw];
  const float* ea = enc + (size_t)a * 64;
  const float* eb = enc + (size_t)b * 64;
  float s = ea[lane] * eb[lane] + ea[lane + 32] * eb[lane + 32];
#pragma unroll
  for (int off = 16; off > 0; off >>= 1)
    s += __shfl_xor(s, off, 32);
  if (lane == 0) out[gw] = 1.0f / (1.0f + __expf(-s));
}

// ---------------------------------------------------------------------------
// Host-side launcher
// ---------------------------------------------------------------------------
extern "C" void kernel_launch(void* const* d_in, const int* in_sizes, int n_in,
                              void* d_out, int out_size, void* d_ws, size_t ws_size,
                              hipStream_t stream) {
  (void)n_in; (void)out_size; (void)ws_size;
  const float* x      = (const float*)d_in[0];   // [N,64]
  const int*   ei     = (const int*)  d_in[1];   // [2,E]
  const float* ew     = (const float*)d_in[2];   // [E]
  const int*   eli    = (const int*)  d_in[3];   // [2,P]
  const float* W_in   = (const float*)d_in[4];   // [64,128]
  const float* b_in   = (const float*)d_in[5];   // [128]
  const float* W_hid  = (const float*)d_in[6];   // [4,128,128]
  const float* b_hid  = (const float*)d_in[7];   // [4,128]
  const float* W_out  = (const float*)d_in[8];   // [128,64]
  const float* b_out  = (const float*)d_in[9];   // [64]

  const int N = in_sizes[0] / 64;
  const int E = in_sizes[2];
  const int P = in_sizes[3] / 2;
  const int* row = ei;
  const int* col = ei + E;

  // Workspace carve-up (256B aligned slices of d_ws)
  char* wp = (char*)d_ws;
  auto carve = [&](size_t bytes) {
    void* p = (void*)wp;
    wp += (bytes + 255) & ~(size_t)255;
    return p;
  };
  float*     dinv = (float*)carve((size_t)N * 4);
  float*     norm = (float*)carve((size_t)E * 4);
  float*     F0   = (float*)carve((size_t)N * 128 * 4);
  float*     F1   = (float*)carve((size_t)N * 128 * 4);
  float*     H    = (float*)carve((size_t)N * 128 * 4);
  _Float16*  Wpk  = (_Float16*)carve((size_t)128 * 128 * 2);  // packed weights

  const int T = 256;

  // ---- gcn_norm (once, shared by all layers) ----
  zero_f32   <<<(N + T - 1) / T, T, 0, stream>>>(dinv, N);
  deg_accum  <<<(E + T - 1) / T, T, 0, stream>>>(col, ew, dinv, E);
  deg_to_dinv<<<(N + T - 1) / T, T, 0, stream>>>(dinv, N);
  edge_norm  <<<(E + T - 1) / T, T, 0, stream>>>(row, col, ew, dinv, norm, E);

  const int nStrips    = N / 16;                 // 6250
  const int gemmBlock  = 64;                     // 2 waves per block
  const int gemmGrid   = (nStrips * 32 + gemmBlock - 1) / gemmBlock;

  // ---- one GCN layer ----
  auto layer = [&](const float* in, int Cin, const float* W, const float* b,
                   int Cout, float* out, int relu) {
    // pack weights to fp16 fragment order (tiny: <= 16384 elements)
    int total = Cin * Cout;
    pack_w_f16<<<(total + T - 1) / T, T, 0, stream>>>(W, Wpk, Cout, Cout / 16, total);
    // WMMA GEMM (templated on Cin/32, Cout/16)
    if (Cin == 64 && Cout == 128)
      gcn_gemm_wmma<2, 8><<<gemmGrid, gemmBlock, 0, stream>>>(in, Wpk, H, nStrips);
    else if (Cin == 128 && Cout == 128)
      gcn_gemm_wmma<4, 8><<<gemmGrid, gemmBlock, 0, stream>>>(in, Wpk, H, nStrips);
    else
      gcn_gemm_wmma<4, 4><<<gemmGrid, gemmBlock, 0, stream>>>(in, Wpk, H, nStrips);
    // aggregate: bias init -> edge scatter -> self-loop (+ReLU)
    size_t tot    = (size_t)N * Cout;
    int    blocks = (int)((tot + T - 1) / T);
    int    cshift = (Cout == 128) ? 7 : 6;
    init_bias<<<blocks, T, 0, stream>>>(out, b, tot, Cout - 1);
    int edgeBlocks = (int)(((size_t)E * 32 + T - 1) / T);
    scatter_edges<<<edgeBlocks, T, 0, stream>>>(row, col, norm, H, out, E, Cout);
    selfloop_epilogue<<<blocks, T, 0, stream>>>(out, H, dinv, tot, cshift, relu);
  };

  layer(x,  64,  W_in,              b_in,          128, F0, 1);
  layer(F0, 128, W_hid + 0 * 16384, b_hid + 0*128, 128, F1, 1);
  layer(F1, 128, W_hid + 1 * 16384, b_hid + 1*128, 128, F0, 1);
  layer(F0, 128, W_hid + 2 * 16384, b_hid + 2*128, 128, F1, 1);
  layer(F1, 128, W_hid + 3 * 16384, b_hid + 3*128, 128, F0, 1);
  layer(F0, 128, W_out,             b_out,         64,  F1, 0);   // enc in F1 [N,64]

  // ---- dot-product link decoder ----
  int pairBlocks = (int)(((size_t)P * 32 + T - 1) / T);
  decode_pairs<<<pairBlocks, T, 0, stream>>>(F1, eli, (float*)d_out, P);
}
